// SwarmGNN_20615843021225
// MI455X (gfx1250) — matrix operand order repaired and checked by reading, not compiled, not run on hardware
//
#include <hip/hip_runtime.h>

typedef __attribute__((ext_vector_type(16))) _Float16 v16h;
typedef __attribute__((ext_vector_type(8)))  _Float16 v8h;
typedef __attribute__((ext_vector_type(4)))  _Float16 v4h;
typedef __attribute__((ext_vector_type(8)))  float    v8f;
typedef __attribute__((ext_vector_type(4)))  float    v4f;

#define HID  128
#define EMB  64
#define CATP 160   // 136 padded to multiple of 32
#define WAVES 8

// ---------------- WMMA helpers (gfx1250 layouts) ----------------

__device__ __forceinline__ v8f wmma_f16(v16h a, v16h b, v8f c) {
  return __builtin_amdgcn_wmma_f32_16x16x32_f16(false, a, false, b, (short)0, c,
                                                false, false);
}

// A fragment (16x32 f16 tile, M=rows): lane<16 -> K 0..7 & 16..23 of row=lane,
// lane>=16 -> K 8..15 & 24..31 of row=lane-16. Tile in LDS row-major [16][lda].
__device__ __forceinline__ v16h ld_a_frag(const _Float16* tile, int lda, int kc, int lane) {
  int r  = lane & 15;
  int kk = kc + ((lane >> 4) << 3);
  const _Float16* p = tile + r * lda + kk;
  v8h lo = *(const v8h*)p;
  v8h hi = *(const v8h*)(p + 16);
  return __builtin_shufflevector(lo, hi, 0,1,2,3,4,5,6,7,8,9,10,11,12,13,14,15);
}

// B fragment (32x16 f16, KxN): lane<16 -> col n=lane, K kc+0..7 & kc+16..23;
// lane>=16 -> col n=lane-16, K kc+8..15 & kc+24..31. Weights in LDS as [N][kpad].
__device__ __forceinline__ v16h ld_b_frag(const _Float16* w, int kpad, int nt, int kc, int lane) {
  int n  = (lane & 15) + (nt << 4);
  int kk = kc + ((lane >> 4) << 3);
  const _Float16* p = w + n * kpad + kk;
  v8h lo = *(const v8h*)p;
  v8h hi = *(const v8h*)(p + 16);
  return __builtin_shufflevector(lo, hi, 0,1,2,3,4,5,6,7,8,9,10,11,12,13,14,15);
}

// ---------------- small kernels ----------------

__global__ __launch_bounds__(256) void k_zero(float* __restrict__ p, int n) {
  int i = blockIdx.x * 256 + threadIdx.x;
  if (i < n) p[i] = 0.f;
}

__global__ __launch_bounds__(256) void k_edgeattr(const float* __restrict__ pos,
                                                  const int* __restrict__ eidx,
                                                  _Float16* __restrict__ ea, int E) {
  int e = blockIdx.x * 256 + threadIdx.x;
  if (e >= E) return;
  int s = eidx[e];
  int d = eidx[E + e];
  float rx = pos[2 * d]     - pos[2 * s];
  float ry = pos[2 * d + 1] - pos[2 * s + 1];
  float dist = sqrtf(rx * rx + ry * ry);
  float inv  = 1.f / (dist + 1e-6f);
  float dn   = dist * (1.f / 1500.f);
  float ang  = atan2f(ry, rx);
  v8h o;
  o[0] = (_Float16)dn;
  o[1] = (_Float16)(rx * inv);
  o[2] = (_Float16)(ry * inv);
  o[3] = (_Float16)cosf(ang);
  o[4] = (_Float16)sinf(ang);
  o[5] = (_Float16)(dn * dn);
  o[6] = (_Float16)0.f;
  o[7] = (_Float16)0.f;
  *(v8h*)(ea + (size_t)e * 8) = o;
}

// encoder: h = relu(LN(obs@W1+b1)); x = relu(h@W2+b2). One node per block.
__global__ __launch_bounds__(128) void k_encode(
    const float* __restrict__ obs, const float* __restrict__ w1,
    const float* __restrict__ b1, const float* __restrict__ g,
    const float* __restrict__ beta, const float* __restrict__ w2,
    const float* __restrict__ b2, float* __restrict__ x,
    _Float16* __restrict__ xh, int N) {
  __shared__ float sh[128];
  __shared__ float red[128];
  __shared__ float ob[25];
  int node = blockIdx.x;
  if (node >= N) return;
  int t = threadIdx.x;
  if (t < 25) ob[t] = obs[(size_t)node * 25 + t];
  __syncthreads();
  float h = b1[t];
  #pragma unroll
  for (int k = 0; k < 25; ++k) h += ob[k] * w1[k * 128 + t];
  red[t] = h;
  __syncthreads();
  for (int s = 64; s > 0; s >>= 1) { if (t < s) red[t] += red[t + s]; __syncthreads(); }
  float mu = red[0] * (1.f / 128.f);
  __syncthreads();
  float dv = (h - mu) * (h - mu);
  red[t] = dv;
  __syncthreads();
  for (int s = 64; s > 0; s >>= 1) { if (t < s) red[t] += red[t + s]; __syncthreads(); }
  float var = red[0] * (1.f / 128.f);
  float y = (h - mu) * rsqrtf(var + 1e-5f) * g[t] + beta[t];
  y = y > 0.f ? y : 0.f;
  sh[t] = y;
  __syncthreads();
  if (t < 64) {
    float acc = b2[t];
    #pragma unroll 8
    for (int k = 0; k < 128; ++k) acc += sh[k] * w2[k * 64 + t];
    acc = acc > 0.f ? acc : 0.f;
    x[(size_t)node * 64 + t]  = acc;
    xh[(size_t)node * 64 + t] = (_Float16)acc;
  }
}

// ---------------- edge kernel: m = relu(cat@W1+b1)@W2+b2; aggr[dst] += 0.25*m ---------

__global__ __launch_bounds__(256) void k_edge(
    const _Float16* __restrict__ xh, const _Float16* __restrict__ ea,
    const int* __restrict__ eidx, int E,
    const float* __restrict__ w1, const float* __restrict__ b1,
    const float* __restrict__ w2, const float* __restrict__ b2,
    float* __restrict__ aggr, int ntiles) {
  extern __shared__ char smem[];
  _Float16* W1 = (_Float16*)smem;                  // [128][160] f16 = 40960 B
  _Float16* W2 = (_Float16*)(smem + 40960);        // [64][128]  f16 = 16384 B
  _Float16* Ab = (_Float16*)(smem + 57344);        // 8 waves * [16][160]
  _Float16* Hb = (_Float16*)(smem + 98304);        // 8 waves * [16][128]

  int tid = threadIdx.x;
  for (int i = tid; i < HID * CATP; i += 256) {
    int n = i / CATP, k = i - n * CATP;
    float v = (k < 136) ? w1[k * HID + n] : 0.f;
    W1[i] = (_Float16)v;
  }
  for (int i = tid; i < EMB * HID; i += 256) {
    int n = i >> 7, k = i & 127;
    W2[i] = (_Float16)w2[k * EMB + n];
  }
  __syncthreads();

  int wid = tid >> 5, lane = tid & 31;
  _Float16* Aw = Ab + wid * (16 * CATP);
  _Float16* Hw = Hb + wid * (16 * HID);
  int wg = blockIdx.x * WAVES + wid;
  int nw = gridDim.x * WAVES;

  for (int tile = wg; tile < ntiles; tile += nw) {
    int e0 = tile * 16;
    // ---- gather cat = [xh[dst], xh[src], edge_attr, 0-pad] into A tile ----
    {
      int e = lane >> 1, hf = lane & 1;
      int eg = e0 + e; if (eg >= E) eg = E - 1;
      int s = eidx[eg];
      int d = eidx[E + eg];
      const v8h* ps = (const v8h*)(xh + (size_t)s * EMB + hf * 32);
      const v8h* pd = (const v8h*)(xh + (size_t)d * EMB + hf * 32);
      v8h* rd = (v8h*)(Aw + e * CATP + hf * 32);
      v8h* rs = (v8h*)(Aw + e * CATP + 64 + hf * 32);
      rd[0] = pd[0]; rd[1] = pd[1]; rd[2] = pd[2]; rd[3] = pd[3];
      rs[0] = ps[0]; rs[1] = ps[1]; rs[2] = ps[2]; rs[3] = ps[3];
      if (lane < 16) {
        int eg2 = e0 + lane; if (eg2 >= E) eg2 = E - 1;
        v8h* ra = (v8h*)(Aw + lane * CATP + 128);
        v8h z = {};
        ra[0] = *(const v8h*)(ea + (size_t)eg2 * 8);
        ra[1] = z; ra[2] = z; ra[3] = z;
      }
    }
    // ---- GEMM1: [16][160] @ [160][128] -> relu -> H ----
    v16h af[5];
    #pragma unroll
    for (int kc = 0; kc < 5; ++kc) af[kc] = ld_a_frag(Aw, CATP, kc * 32, lane);
    int rb = (lane >> 4) << 3;
    #pragma unroll
    for (int nt = 0; nt < 8; ++nt) {
      v8f acc = {};
      #pragma unroll
      for (int kc = 0; kc < 5; ++kc)
        acc = wmma_f16(af[kc], ld_b_frag(W1, CATP, nt, kc * 32, lane), acc);
      int n = (lane & 15) + (nt << 4);
      float bias = b1[n];
      #pragma unroll
      for (int v = 0; v < 8; ++v) {
        float hv = acc[v] + bias;
        hv = hv > 0.f ? hv : 0.f;
        Hw[(rb + v) * HID + n] = (_Float16)hv;
      }
    }
    // ---- GEMM2: [16][128] @ [128][64] -> 0.25*(.+b2) atomically into aggr[dst] ----
    v16h hf2[4];
    #pragma unroll
    for (int kc = 0; kc < 4; ++kc) hf2[kc] = ld_a_frag(Hw, HID, kc * 32, lane);
    int didx[8];
    #pragma unroll
    for (int v = 0; v < 8; ++v) {
      int eg = e0 + rb + v;
      didx[v] = (eg < E) ? eidx[E + eg] : -1;
    }
    #pragma unroll
    for (int nt = 0; nt < 4; ++nt) {
      v8f acc = {};
      #pragma unroll
      for (int kc = 0; kc < 4; ++kc)
        acc = wmma_f16(hf2[kc], ld_b_frag(W2, HID, nt, kc * 32, lane), acc);
      int n = (lane & 15) + (nt << 4);
      float bias = b2[n];
      #pragma unroll
      for (int v = 0; v < 8; ++v) {
        if (didx[v] >= 0) {
          float m = (acc[v] + bias) * 0.25f;  // softmax-mean == exactly 1/4
          atomicAdd(aggr + (size_t)didx[v] * EMB + n, m);
        }
      }
    }
  }
}

// ---------------- node update: x = LN(x + relu([x,aggr]@W1+b1)@W2+b2) ---------------

__global__ __launch_bounds__(256) void k_update(
    float* __restrict__ x, const float* __restrict__ aggr,
    const float* __restrict__ w1, const float* __restrict__ b1,
    const float* __restrict__ w2, const float* __restrict__ b2,
    const float* __restrict__ g, const float* __restrict__ bb,
    _Float16* __restrict__ xh, int N, int ntiles) {
  extern __shared__ char smem[];
  _Float16* W1 = (_Float16*)smem;                  // [128][128] = 32768 B
  _Float16* W2 = (_Float16*)(smem + 32768);        // [64][128]  = 16384 B
  _Float16* Ab = (_Float16*)(smem + 49152);        // 8 * [16][128]
  _Float16* Hb = (_Float16*)(smem + 81920);        // 8 * [16][128]
  float*    Sb = (float*)   (smem + 114688);       // 8 * [16][64] f32

  int tid = threadIdx.x;
  for (int i = tid; i < HID * HID; i += 256) {
    int n = i >> 7, k = i & 127;
    W1[i] = (_Float16)w1[k * HID + n];
  }
  for (int i = tid; i < EMB * HID; i += 256) {
    int n = i >> 7, k = i & 127;
    W2[i] = (_Float16)w2[k * EMB + n];
  }
  __syncthreads();

  int wid = tid >> 5, lane = tid & 31;
  _Float16* Aw = Ab + wid * (16 * HID);
  _Float16* Hw = Hb + wid * (16 * HID);
  float*    Sw = Sb + wid * (16 * EMB);
  int wg = blockIdx.x * WAVES + wid;
  int nw = gridDim.x * WAVES;

  for (int tile = wg; tile < ntiles; tile += nw) {
    int t0 = tile * 16;
    // gather [x, aggr] -> f16 A tile
    {
      int e = lane >> 1, hf = lane & 1;
      int node = t0 + e; if (node >= N) node = N - 1;
      const float* px = x    + (size_t)node * EMB + hf * 32;
      const float* pa = aggr + (size_t)node * EMB + hf * 32;
      _Float16* r0 = Aw + e * HID + hf * 32;
      _Float16* r1 = Aw + e * HID + 64 + hf * 32;
      #pragma unroll
      for (int j = 0; j < 32; j += 4) {
        v4f vx = *(const v4f*)(px + j);
        v4f va = *(const v4f*)(pa + j);
        v4h hx, ha;
        hx[0] = (_Float16)vx[0]; hx[1] = (_Float16)vx[1];
        hx[2] = (_Float16)vx[2]; hx[3] = (_Float16)vx[3];
        ha[0] = (_Float16)va[0]; ha[1] = (_Float16)va[1];
        ha[2] = (_Float16)va[2]; ha[3] = (_Float16)va[3];
        *(v4h*)(r0 + j) = hx;
        *(v4h*)(r1 + j) = ha;
      }
    }
    v16h af[4];
    #pragma unroll
    for (int kc = 0; kc < 4; ++kc) af[kc] = ld_a_frag(Aw, HID, kc * 32, lane);
    int rb = (lane >> 4) << 3;
    #pragma unroll
    for (int nt = 0; nt < 8; ++nt) {
      v8f acc = {};
      #pragma unroll
      for (int kc = 0; kc < 4; ++kc)
        acc = wmma_f16(af[kc], ld_b_frag(W1, HID, nt, kc * 32, lane), acc);
      int n = (lane & 15) + (nt << 4);
      float bias = b1[n];
      #pragma unroll
      for (int v = 0; v < 8; ++v) {
        float hv = acc[v] + bias;
        hv = hv > 0.f ? hv : 0.f;
        Hw[(rb + v) * HID + n] = (_Float16)hv;
      }
    }
    v16h hf2[4];
    #pragma unroll
    for (int kc = 0; kc < 4; ++kc) hf2[kc] = ld_a_frag(Hw, HID, kc * 32, lane);
    #pragma unroll
    for (int nt = 0; nt < 4; ++nt) {
      v8f acc = {};
      #pragma unroll
      for (int kc = 0; kc < 4; ++kc)
        acc = wmma_f16(hf2[kc], ld_b_frag(W2, HID, nt, kc * 32, lane), acc);
      int n = (lane & 15) + (nt << 4);
      float bias = b2[n];
      #pragma unroll
      for (int v = 0; v < 8; ++v) {
        int row = rb + v, node = t0 + row;
        float res = (node < N) ? x[(size_t)node * EMB + n] : 0.f;
        Sw[row * EMB + n] = res + acc[v] + bias;
      }
    }
    // LayerNorm over 64 features (one row per lane, lanes 0..15)
    if (lane < 16) {
      int node = t0 + lane;
      if (node < N) {
        const float* srow = Sw + lane * EMB;
        float mu = 0.f;
        #pragma unroll
        for (int j = 0; j < EMB; ++j) mu += srow[j];
        mu *= (1.f / 64.f);
        float var = 0.f;
        #pragma unroll
        for (int j = 0; j < EMB; ++j) { float d = srow[j] - mu; var += d * d; }
        var *= (1.f / 64.f);
        float inv = rsqrtf(var + 1e-5f);
        #pragma unroll
        for (int j = 0; j < EMB; ++j) {
          float y = (srow[j] - mu) * inv * g[j] + bb[j];
          x[(size_t)node * EMB + j]  = y;
          xh[(size_t)node * EMB + j] = (_Float16)y;
        }
      }
    }
  }
}

// ---------------- output projection: relu(x@W1+b1)@W2+b2 ----------------

__global__ __launch_bounds__(256) void k_out(
    const float* __restrict__ x,
    const float* __restrict__ w1, const float* __restrict__ b1,
    const float* __restrict__ w2, const float* __restrict__ b2,
    float* __restrict__ out, int N, int ntiles) {
  extern __shared__ char smem[];
  _Float16* W1 = (_Float16*)smem;                  // [128][64] = 16384 B
  _Float16* W2 = (_Float16*)(smem + 16384);        // [64][128] = 16384 B
  _Float16* Ab = (_Float16*)(smem + 32768);        // 8 * [16][64]
  _Float16* Hb = (_Float16*)(smem + 49152);        // 8 * [16][128]

  int tid = threadIdx.x;
  for (int i = tid; i < HID * EMB; i += 256) {
    int n = i >> 6, k = i & 63;
    W1[i] = (_Float16)w1[k * HID + n];             // out_w1 [64,128]
  }
  for (int i = tid; i < EMB * HID; i += 256) {
    int n = i >> 7, k = i & 127;
    W2[i] = (_Float16)w2[k * EMB + n];             // out_w2 [128,64]
  }
  __syncthreads();

  int wid = tid >> 5, lane = tid & 31;
  _Float16* Aw = Ab + wid * (16 * EMB);
  _Float16* Hw = Hb + wid * (16 * HID);
  int wg = blockIdx.x * WAVES + wid;
  int nw = gridDim.x * WAVES;

  for (int tile = wg; tile < ntiles; tile += nw) {
    int t0 = tile * 16;
    {
      int e = lane >> 1, hf = lane & 1;
      int node = t0 + e; if (node >= N) node = N - 1;
      const float* px = x + (size_t)node * EMB + hf * 32;
      _Float16* r0 = Aw + e * EMB + hf * 32;
      #pragma unroll
      for (int j = 0; j < 32; j += 4) {
        v4f vx = *(const v4f*)(px + j);
        v4h hx;
        hx[0] = (_Float16)vx[0]; hx[1] = (_Float16)vx[1];
        hx[2] = (_Float16)vx[2]; hx[3] = (_Float16)vx[3];
        *(v4h*)(r0 + j) = hx;
      }
    }
    v16h af[2];
    #pragma unroll
    for (int kc = 0; kc < 2; ++kc) af[kc] = ld_a_frag(Aw, EMB, kc * 32, lane);
    int rb = (lane >> 4) << 3;
    #pragma unroll
    for (int nt = 0; nt < 8; ++nt) {
      v8f acc = {};
      #pragma unroll
      for (int kc = 0; kc < 2; ++kc)
        acc = wmma_f16(af[kc], ld_b_frag(W1, EMB, nt, kc * 32, lane), acc);
      int n = (lane & 15) + (nt << 4);
      float bias = b1[n];
      #pragma unroll
      for (int v = 0; v < 8; ++v) {
        float hv = acc[v] + bias;
        hv = hv > 0.f ? hv : 0.f;
        Hw[(rb + v) * HID + n] = (_Float16)hv;
      }
    }
    v16h hf2[4];
    #pragma unroll
    for (int kc = 0; kc < 4; ++kc) hf2[kc] = ld_a_frag(Hw, HID, kc * 32, lane);
    #pragma unroll
    for (int nt = 0; nt < 4; ++nt) {
      v8f acc = {};
      #pragma unroll
      for (int kc = 0; kc < 4; ++kc)
        acc = wmma_f16(hf2[kc], ld_b_frag(W2, HID, nt, kc * 32, lane), acc);
      int n = (lane & 15) + (nt << 4);
      float bias = b2[n];
      #pragma unroll
      for (int v = 0; v < 8; ++v) {
        int node = t0 + rb + v;
        if (node < N) out[(size_t)node * EMB + n] = acc[v] + bias;
      }
    }
  }
}

// ---------------- host driver ----------------

extern "C" void kernel_launch(void* const* d_in, const int* in_sizes, int n_in,
                              void* d_out, int out_size, void* d_ws, size_t ws_size,
                              hipStream_t stream) {
  (void)n_in; (void)out_size; (void)ws_size;
  const float* obs    = (const float*)d_in[0];
  const float* pos    = (const float*)d_in[1];
  const int*   eidx   = (const int*)d_in[2];
  const float* enc_w1 = (const float*)d_in[3];
  const float* enc_b1 = (const float*)d_in[4];
  const float* enc_g  = (const float*)d_in[5];
  const float* enc_b  = (const float*)d_in[6];
  const float* enc_w2 = (const float*)d_in[7];
  const float* enc_b2 = (const float*)d_in[8];
  const float* msg_w1 = (const float*)d_in[9];
  const float* msg_b1 = (const float*)d_in[10];
  const float* msg_w2 = (const float*)d_in[11];
  const float* msg_b2 = (const float*)d_in[12];
  // d_in[13..16] (attention) unused: mean(softmax(a)) == 0.25 exactly.
  const float* upd_w1 = (const float*)d_in[17];
  const float* upd_b1 = (const float*)d_in[18];
  const float* upd_w2 = (const float*)d_in[19];
  const float* upd_b2 = (const float*)d_in[20];
  const float* ln_g   = (const float*)d_in[21];
  const float* ln_b   = (const float*)d_in[22];
  const float* out_w1 = (const float*)d_in[23];
  const float* out_b1 = (const float*)d_in[24];
  const float* out_w2 = (const float*)d_in[25];
  const float* out_b2 = (const float*)d_in[26];

  int N = in_sizes[0] / 25;
  int E = in_sizes[2] / 2;

  char* p = (char*)d_ws;
  auto carve = [&](size_t bytes) -> char* {
    char* r = p;
    p += (bytes + 255) & ~(size_t)255;
    return r;
  };
  float*    x    = (float*)carve((size_t)N * 64 * 4);
  float*    aggr = (float*)carve((size_t)N * 64 * 4);
  _Float16* xh   = (_Float16*)carve((size_t)N * 64 * 2);
  _Float16* ea   = (_Float16*)carve((size_t)E * 8 * 2);

  k_encode<<<N, 128, 0, stream>>>(obs, enc_w1, enc_b1, enc_g, enc_b,
                                  enc_w2, enc_b2, x, xh, N);
  k_edgeattr<<<(E + 255) / 256, 256, 0, stream>>>(pos, eidx, ea, E);

  int etiles = (E + 15) / 16;
  int ntile  = (N + 15) / 16;
  int eblocks = (etiles + WAVES - 1) / WAVES; if (eblocks > 1024) eblocks = 1024;
  int nblocks = (ntile + WAVES - 1) / WAVES;  if (nblocks > 1024) nblocks = 1024;

  const size_t SMEM_EDGE = 131072; // W1 40960 + W2 16384 + A 40960 + H 32768
  const size_t SMEM_UPD  = 147456; // W1 32768 + W2 16384 + A 32768 + H 32768 + S 32768
  const size_t SMEM_OUT  = 81920;  // W1 16384 + W2 16384 + A 16384 + H 32768

  for (int l = 0; l < 3; ++l) {
    k_zero<<<(N * 64 + 255) / 256, 256, 0, stream>>>(aggr, N * 64);
    k_edge<<<eblocks, 256, SMEM_EDGE, stream>>>(
        xh, ea, eidx, E,
        msg_w1 + (size_t)l * 136 * 128, msg_b1 + (size_t)l * 128,
        msg_w2 + (size_t)l * 128 * 64,  msg_b2 + (size_t)l * 64,
        aggr, etiles);
    k_update<<<nblocks, 256, SMEM_UPD, stream>>>(
        x, aggr,
        upd_w1 + (size_t)l * 128 * 128, upd_b1 + (size_t)l * 128,
        upd_w2 + (size_t)l * 128 * 64,  upd_b2 + (size_t)l * 64,
        ln_g + (size_t)l * 64, ln_b + (size_t)l * 64,
        xh, N, ntile);
  }
  k_out<<<nblocks, 256, SMEM_OUT, stream>>>(x, out_w1, out_b1, out_w2, out_b2,
                                            (float*)d_out, N, ntile);
}
